// GatbertSelfAttention_6322191860303
// MI455X (gfx1250) — compile-verified, matrix operand order checked by reading, not run on hardware
//
#include <hip/hip_runtime.h>
#include <hip/hip_bf16.h>

// Problem constants (match reference)
#define BB      4
#define NN      4096
#define HH      256
#define NHEADS  8
#define DHD     32
#define DEG     16
#define NE      (BB * NN * DEG)          // 262144 edges
#define NROWS   (BB * NN)                // 16384 node rows

#define LDS_ROW 264                      // 256 + 8 ushort pad (16B) -> bank rotation

typedef __attribute__((ext_vector_type(16))) __bf16  v16bf;
typedef __attribute__((ext_vector_type(8)))  float   v8f;
typedef __attribute__((ext_vector_type(4)))  unsigned int v4u;

__device__ __forceinline__ unsigned short f32_bf16_rne(float f) {
    unsigned u = __builtin_bit_cast(unsigned, f);
    unsigned r = u + 0x7FFFu + ((u >> 16) & 1u);   // round to nearest even
    return (unsigned short)(r >> 16);
}

// ---------------- Phase 0a: fp32 -> bf16 convert of node_states (vectorized)
__global__ void cvt_x4_kernel(const float4* __restrict__ x,
                              ushort4* __restrict__ o, int n4) {
    int i = blockIdx.x * blockDim.x + threadIdx.x;
    if (i < n4) {
        float4 f = x[i];
        ushort4 r;
        r.x = f32_bf16_rne(f.x);
        r.y = f32_bf16_rne(f.y);
        r.z = f32_bf16_rne(f.z);
        r.w = f32_bf16_rne(f.w);
        o[i] = r;
    }
}

// ---------------- Phase 0b: transpose + convert W{q,k,v} -> bf16 Wt[z][n][k]
__global__ void cvt_w_kernel(const float* __restrict__ wq,
                             const float* __restrict__ wk,
                             const float* __restrict__ wv,
                             unsigned short* __restrict__ wt) {
    int i = blockIdx.x * blockDim.x + threadIdx.x;
    if (i >= 3 * HH * HH) return;
    int z = i >> 16;            // which matrix
    int r = i & 0xFFFF;
    int n = r >> 8;             // output channel
    int k = r & 255;            // input channel
    const float* w = (z == 0) ? wq : (z == 1) ? wk : wv;
    wt[i] = f32_bf16_rne(w[k * HH + n]);   // Wt[n][k] = W[k][n]
}

// ---------------- Phase 1: fused QKV GEMM via v_wmma_f32_16x16x32_bf16
// Block = 4 waves, all sharing one 64-column weight strip. The 32 KB strip is
// staged global->LDS with gfx1250 async copies (ASYNCcnt), then each wave
// computes a 16(M) x 64(N) tile: A from global (b128), B from LDS (ds b128).
// K = 256 -> 8 wmma k-steps. Grid: (M/16/4, N/64, 3).
__global__ __launch_bounds__(128)
void qkv_gemm_wmma(const unsigned short* __restrict__ xb,   // (16384, 256) bf16
                   const unsigned short* __restrict__ wt,   // (3, 256, 256) bf16, [n][k]
                   const float* __restrict__ bq, const float* __restrict__ bk,
                   const float* __restrict__ bv,
                   float* __restrict__ Q, float* __restrict__ K,
                   float* __restrict__ V) {
    __shared__ __align__(16) unsigned short wlds[64 * LDS_ROW];  // 33 KB

    const int tid  = threadIdx.x;
    const int lane = tid & 31;
    const int wid  = tid >> 5;                 // 4 waves / block
    const int m0   = (blockIdx.x * 4 + wid) * 16;
    const int n0   = blockIdx.y * 64;
    const int z    = blockIdx.z;

    const unsigned short* wstrip = wt + (size_t)z * HH * HH + (size_t)n0 * HH;
    const float* bias = (z == 0) ? bq : (z == 1) ? bk : bv;
    float* out        = (z == 0) ? Q  : (z == 1) ? K  : V;

    // ---- async stage: 64 rows x 512 B -> LDS rows padded to 528 B ----
    {
        unsigned base = (unsigned)(size_t)(&wlds[0]);   // low 32 bits = LDS addr
        #pragma unroll
        for (int c = tid; c < 2048; c += 128) {         // 2048 x 16 B chunks
            int row = c >> 5;
            int off = c & 31;
            unsigned ldsa = base + (unsigned)(row * (LDS_ROW * 2) + off * 16);
            const unsigned short* gp = wstrip + row * HH + off * 8;
            asm volatile("global_load_async_to_lds_b128 %0, %1, off"
                         :: "v"(ldsa), "v"(gp) : "memory");
        }
        asm volatile("s_wait_asynccnt 0" ::: "memory");
        __syncthreads();   // other waves' async copies also complete-before-signal
    }

    const int col = lane & 15;     // M-row for A, N-col for B/C/D
    const int kg  = lane >> 4;     // K sub-group select per ISA layout

    v8f acc[4];
    for (int t = 0; t < 4; ++t)
        acc[t] = (v8f){0.f, 0.f, 0.f, 0.f, 0.f, 0.f, 0.f, 0.f};

    for (int kk = 0; kk < HH; kk += 32) {
        // A fragment: lane holds row (lane&15); halves 0..7 = K kk+kg*8..+7,
        // halves 8..15 = K kk+16+kg*8..+7  (two b128 loads)
        union { v16bf v; v4u q[2]; } a;
        const unsigned short* ap = xb + (size_t)(m0 + col) * HH + kk + kg * 8;
        a.q[0] = *(const v4u*)(ap);
        a.q[1] = *(const v4u*)(ap + 16);

        #pragma unroll
        for (int t = 0; t < 4; ++t) {
            union { v16bf v; v4u q[2]; } b;
            const unsigned short* bp =
                &wlds[(t * 16 + col) * LDS_ROW + kk + kg * 8];
            b.q[0] = *(const v4u*)(bp);
            b.q[1] = *(const v4u*)(bp + 16);
            acc[t] = __builtin_amdgcn_wmma_f32_16x16x32_bf16(
                false, a.v, false, b.v, (short)0, acc[t], false, false);
        }
    }

    // C/D layout: col = lane&15, row r -> M = r + 8*(lane>>4)
    for (int t = 0; t < 4; ++t) {
        int n = n0 + t * 16 + col;
        float bsn = bias[n];
        float* op = out + (size_t)(m0 + 8 * kg) * HH + n;
        #pragma unroll
        for (int r = 0; r < 8; ++r)
            op[(size_t)r * HH] = acc[t][r] + bsn;
    }
}

// ---------------- Phase 2: per-target edge attention (one wave per target)
// Lane L owns 8 contiguous channels L*8..L*8+7 => head = L>>2.
// Per-head score = 4-lane butterfly reduce; online softmax over 16 edges.
__global__ __launch_bounds__(256)
void edge_attn_kernel(const float* __restrict__ Q, const float* __restrict__ K,
                      const float* __restrict__ V, const int* __restrict__ edges,
                      float* __restrict__ out) {
    const int lane = threadIdx.x & 31;
    const int wid  = threadIdx.x >> 5;
    const int seg  = blockIdx.x * 8 + wid;       // (b*N + target), 16384 total
    if (seg >= NROWS) return;
    const int b = seg >> 12;                     // N = 4096
    const size_t qoff = (size_t)seg * HH;

    // load this target's q channels (2 x b128)
    const float4* qp = (const float4*)(Q + qoff) + lane * 2;
    float4 q0 = qp[0], q1 = qp[1];

    float m = -1e30f, l = 0.f;
    float4 a0 = {0.f, 0.f, 0.f, 0.f}, a1 = {0.f, 0.f, 0.f, 0.f};
    const float scale = 0.17677669529663687f;    // 1/sqrt(32)
    const int* srcp = edges + 2 * NE + seg * DEG;

    #pragma unroll 4
    for (int e = 0; e < DEG; ++e) {
        int src = srcp[e];
        size_t off = ((size_t)(b * NN + src)) * HH;

        const float4* kp = (const float4*)(K + off) + lane * 2;
        float4 k0 = kp[0], k1 = kp[1];
        float p = q0.x * k0.x + q0.y * k0.y + q0.z * k0.z + q0.w * k0.w
                + q1.x * k1.x + q1.y * k1.y + q1.z * k1.z + q1.w * k1.w;
        // reduce across the 4 lanes of this head (lanes 4j..4j+3)
        p += __shfl_xor(p, 1, 32);
        p += __shfl_xor(p, 2, 32);
        float s = p * scale;

        const float4* vp = (const float4*)(V + off) + lane * 2;
        float4 v0 = vp[0], v1 = vp[1];

        float mn  = fmaxf(m, s);
        float sc  = __expf(m - mn);
        float wgt = __expf(s - mn);
        l = l * sc + wgt;
        a0.x = a0.x * sc + wgt * v0.x;  a0.y = a0.y * sc + wgt * v0.y;
        a0.z = a0.z * sc + wgt * v0.z;  a0.w = a0.w * sc + wgt * v0.w;
        a1.x = a1.x * sc + wgt * v1.x;  a1.y = a1.y * sc + wgt * v1.y;
        a1.z = a1.z * sc + wgt * v1.z;  a1.w = a1.w * sc + wgt * v1.w;
        m = mn;
    }

    float inv = 1.0f / l;
    float4 o0 = {a0.x * inv, a0.y * inv, a0.z * inv, a0.w * inv};
    float4 o1 = {a1.x * inv, a1.y * inv, a1.z * inv, a1.w * inv};
    float4* op = (float4*)(out + qoff) + lane * 2;
    op[0] = o0;
    op[1] = o1;
}

extern "C" void kernel_launch(void* const* d_in, const int* in_sizes, int n_in,
                              void* d_out, int out_size, void* d_ws, size_t ws_size,
                              hipStream_t stream) {
    const float* x     = (const float*)d_in[0];
    const int*   edges = (const int*)d_in[1];
    const float* Wq    = (const float*)d_in[2];
    const float* bq    = (const float*)d_in[3];
    const float* Wk    = (const float*)d_in[4];
    const float* bk    = (const float*)d_in[5];
    const float* Wv    = (const float*)d_in[6];
    const float* bv    = (const float*)d_in[7];
    float* out = (float*)d_out;

    // Workspace carve-out (all offsets 256B-aligned):
    //   xb  : bf16 node_states            8 MiB
    //   wt  : bf16 W^T (3,256,256)        384 KiB (padded to 512 KiB)
    //   Q/K/V : fp32 projections          3 x 16 MiB
    char* ws = (char*)d_ws;
    unsigned short* xb = (unsigned short*)ws;
    unsigned short* wt = (unsigned short*)(ws + (size_t)8 * 1024 * 1024);
    float* Qf = (float*)(ws + (size_t)8 * 1024 * 1024 + 512 * 1024);
    float* Kf = Qf + (size_t)NROWS * HH;
    float* Vf = Kf + (size_t)NROWS * HH;

    // Phase 0: precision conversion
    int n4 = NROWS * HH / 4;
    cvt_x4_kernel<<<(n4 + 255) / 256, 256, 0, stream>>>(
        (const float4*)x, (ushort4*)xb, n4);
    cvt_w_kernel<<<(3 * HH * HH + 255) / 256, 256, 0, stream>>>(Wq, Wk, Wv, wt);

    // Phase 1: QKV projections via WMMA (weights async-staged to LDS)
    dim3 g(NROWS / 16 / 4, HH / 64, 3);   // (256, 4, 3)
    qkv_gemm_wmma<<<g, 128, 0, stream>>>(xb, wt, bq, bk, bv, Qf, Kf, Vf);

    // Phase 2: per-target-node edge softmax attention
    edge_attn_kernel<<<NROWS / 8, 256, 0, stream>>>(Qf, Kf, Vf, edges, out);
}
